// GATLayer_32993938767998
// MI455X (gfx1250) — matrix operand order repaired and checked by reading, not compiled, not run on hardware
//
#include <hip/hip_runtime.h>
#include <hip/hip_bf16.h>

typedef __attribute__((ext_vector_type(16))) _Float16 v16h;
typedef __attribute__((ext_vector_type(8)))  _Float16 v8h;
typedef __attribute__((ext_vector_type(8)))  float    v8f;

#define NV 8192
#define BV 4
#define MTOT (BV * NV)      // 32768 vertices
#define DD 128
#define KN 16               // neighbors
#define SS 17               // K+1 attention positions

// ---------------------------------------------------------------------------
// Kernel 1: convert six 128x128 fp32 weights into f16 fragments pre-swizzled
// into this file's B-fragment lane layout (16x16x32 f16, contiguous k).
// Fragment layout (halfs): [w][kb(4)][nb(8)][lane(32)][elem(16)]
//   lane hi = lane>>4 selects K half (0..15 vs 16..31 within the 32-chunk),
//   elem 2v+p holds W[kb*32 + hi*16 + 2v + p][nb*16 + (lane&15)].
// ---------------------------------------------------------------------------
__global__ void prep_weights_kernel(const float* __restrict__ Wq,
                                    const float* __restrict__ Wk,
                                    const float* __restrict__ Wv,
                                    const float* __restrict__ Wo,
                                    const float* __restrict__ W1,
                                    const float* __restrict__ W2,
                                    _Float16* __restrict__ frag) {
    int chunk = blockIdx.x;          // 0..31 : kb*8 + nb
    int w     = blockIdx.y;          // 0..5
    int lane  = threadIdx.x & 31;
    int kb = chunk >> 3;
    int nb = chunk & 7;
    int hi = lane >> 4;
    int c  = lane & 15;

    const float* W;
    switch (w) {
        case 0: W = Wq; break;
        case 1: W = Wk; break;
        case 2: W = Wv; break;
        case 3: W = Wo; break;
        case 4: W = W1; break;
        default: W = W2; break;
    }

    _Float16* dst = frag + ((size_t)(w * 32 + chunk) * 32 + lane) * 16;
    int n = nb * 16 + c;
    #pragma unroll
    for (int v = 0; v < 8; ++v) {
        int k0 = kb * 32 + hi * 16 + 2 * v;
        dst[2 * v]     = (_Float16)W[(size_t)k0 * DD + n];
        dst[2 * v + 1] = (_Float16)W[(size_t)(k0 + 1) * DD + n];
    }
}

// ---------------------------------------------------------------------------
// Kernel 2: generic M x 128 @ 128 x 128 GEMM, fp32 in, f16 WMMA, fp32 acc.
// Block = 256 threads = 8 waves; block owns one 16-row m-tile, wave w owns
// n-tile w. The fp32 A tile is staged ONCE per block into LDS as f16 in the
// contiguous-k A-fragment layout matching B, then every wave feeds its 4 WMMAs from
// two ds_load_b128 per k-chunk. bias[n] added; optional ReLU.
// ---------------------------------------------------------------------------
__global__ void gemm128_kernel(const float* __restrict__ A,
                               const _Float16* __restrict__ Bfrag,
                               const float* __restrict__ bias,
                               float* __restrict__ C,
                               int relu) {
    __shared__ _Float16 smem[4 * 32 * 16];   // [kb][lane][16 halfs] = 4 KB

    int tid  = threadIdx.x;
    int lane = tid & 31;
    int wv   = tid >> 5;                     // n-tile 0..7
    int m0   = blockIdx.x * 16;

    // ---- cooperative stage: 16 rows x 16 K-octets, one octet per thread ----
    {
        int r = tid >> 4;                    // row 0..15
        int o = tid & 15;                    // K-octet 0..15 (K = o*8..o*8+7)
        const float* src = A + (size_t)(m0 + r) * DD + o * 8;
        float4 x0 = *(const float4*)(src);
        float4 x1 = *(const float4*)(src + 4);
        v8h p;
        p[0] = (_Float16)x0.x; p[1] = (_Float16)x0.y;
        p[2] = (_Float16)x0.z; p[3] = (_Float16)x0.w;
        p[4] = (_Float16)x1.x; p[5] = (_Float16)x1.y;
        p[6] = (_Float16)x1.z; p[7] = (_Float16)x1.w;
        int kb  = o >> 2;
        int hi  = (o >> 1) & 1;
        int grp = o & 1;
        int slane = hi * 16 + r;
        *(v8h*)(smem + ((size_t)(kb * 32 + slane) * 2 + grp) * 8) = p;
    }
    __syncthreads();

    // ---- WMMA main loop: A from LDS, pre-swizzled B from global ----
    v8f acc = {};
    #pragma unroll
    for (int kb = 0; kb < 4; ++kb) {
        v16h af = *(const v16h*)(smem + (size_t)(kb * 32 + lane) * 16);
        v16h bf = *(const v16h*)(Bfrag + ((size_t)(kb * 8 + wv) * 32 + lane) * 16);
        acc = __builtin_amdgcn_wmma_f32_16x16x32_f16(
            false, af, false, bf, (short)0, acc, false, false);
    }

    // ---- epilogue: bias (+ReLU), strided column store ----
    int hi = lane >> 4;
    int n  = wv * 16 + (lane & 15);
    float bval = bias[n];
    float* crow = C + (size_t)(m0 + hi * 8) * DD + n;
    #pragma unroll
    for (int v = 0; v < 8; ++v) {
        float x = acc[v] + bval;
        if (relu) x = fmaxf(x, 0.0f);
        crow[(size_t)v * DD] = x;
    }
}

// ---------------------------------------------------------------------------
// Kernel 3: attention. One wave per vertex. Lane = head(4) x sub(8),
// each lane owns 4 features of its head (head dim 32). Scores via 4-wide
// dot + xor-shuffle reduction over the 8 lanes of the head group.
// ---------------------------------------------------------------------------
__global__ void attn_kernel(const float* __restrict__ Q,
                            const float* __restrict__ Kh,
                            const float* __restrict__ Vh,
                            const int*   __restrict__ nidx,
                            const int*   __restrict__ vlen,
                            float* __restrict__ Ctx) {
    int wid  = blockIdx.x * (blockDim.x >> 5) + (threadIdx.x >> 5); // vertex
    int lane = threadIdx.x & 31;
    int m = wid;
    int b = m >> 13;            // /8192
    int h   = lane >> 3;
    int sub = lane & 7;
    int fo  = h * 32 + sub * 4; // feature offset

    float4 q4 = *(const float4*)(Q + (size_t)m * DD + fo);
    int vl = vlen[m];

    float scores[SS];
    float4 vr[SS];
    #pragma unroll
    for (int s = 0; s < SS; ++s) {
        int row = (s == 0) ? m : (b * NV + nidx[(size_t)m * KN + (s - 1)]);
        float4 k4 = *(const float4*)(Kh + (size_t)row * DD + fo);
        vr[s]     = *(const float4*)(Vh + (size_t)row * DD + fo);
        float p = q4.x * k4.x + q4.y * k4.y + q4.z * k4.z + q4.w * k4.w;
        p += __shfl_xor(p, 1, 32);
        p += __shfl_xor(p, 2, 32);
        p += __shfl_xor(p, 4, 32);
        float sc = p * 0.17677669529663687f;   // 1/sqrt(32)
        if (s > vl) sc += -1.0e9f;             // mask: pos < valid_len+1
        scores[s] = sc;
    }

    float mx = scores[0];
    #pragma unroll
    for (int s = 1; s < SS; ++s) mx = fmaxf(mx, scores[s]);
    float sum = 0.0f;
    #pragma unroll
    for (int s = 0; s < SS; ++s) { scores[s] = __expf(scores[s] - mx); sum += scores[s]; }
    float inv = 1.0f / sum;

    float4 c = make_float4(0.f, 0.f, 0.f, 0.f);
    #pragma unroll
    for (int s = 0; s < SS; ++s) {
        float w = scores[s] * inv;
        c.x += w * vr[s].x; c.y += w * vr[s].y;
        c.z += w * vr[s].z; c.w += w * vr[s].w;
    }
    *(float4*)(Ctx + (size_t)m * DD + fo) = c;
}

// ---------------------------------------------------------------------------
// Launch
// inputs: 0 vertex_feat, 1 neighbors_idx, 2 valid_lens,
//         3 Wq, 4 bq, 5 Wk, 6 bk, 7 Wv, 8 bv, 9 Wo, 10 bo,
//         11 W1, 12 b1, 13 W2, 14 b2
// ---------------------------------------------------------------------------
extern "C" void kernel_launch(void* const* d_in, const int* in_sizes, int n_in,
                              void* d_out, int out_size, void* d_ws, size_t ws_size,
                              hipStream_t stream) {
    const float* X    = (const float*)d_in[0];
    const int*   nidx = (const int*)d_in[1];
    const int*   vlen = (const int*)d_in[2];
    const float* Wq = (const float*)d_in[3];  const float* bq = (const float*)d_in[4];
    const float* Wk = (const float*)d_in[5];  const float* bk = (const float*)d_in[6];
    const float* Wv = (const float*)d_in[7];  const float* bv = (const float*)d_in[8];
    const float* Wo = (const float*)d_in[9];  const float* bo = (const float*)d_in[10];
    const float* W1 = (const float*)d_in[11]; const float* b1 = (const float*)d_in[12];
    const float* W2 = (const float*)d_in[13]; const float* b2 = (const float*)d_in[14];

    // workspace layout
    char* ws = (char*)d_ws;
    _Float16* frag = (_Float16*)ws;                     // 6 * 32 KB
    const size_t MAT_BYTES  = (size_t)MTOT * DD * 4;    // 16 MB
    float* Qout = (float*)(ws + 262144);
    float* Kout = (float*)(ws + 262144 + MAT_BYTES);
    float* Vout = (float*)(ws + 262144 + 2 * MAT_BYTES);
    float* Ctx  = (float*)(ws + 262144 + 3 * MAT_BYTES);
    float* buf1 = Qout;   // reuse after attention
    float* buf2 = Kout;   // reuse after attention
    (void)in_sizes; (void)n_in; (void)out_size; (void)ws_size;

    const _Float16* fWq = frag + 0 * 16384;
    const _Float16* fWk = frag + 1 * 16384;
    const _Float16* fWv = frag + 2 * 16384;
    const _Float16* fWo = frag + 3 * 16384;
    const _Float16* fW1 = frag + 4 * 16384;
    const _Float16* fW2 = frag + 5 * 16384;

    // 1) weight conversion / swizzle
    prep_weights_kernel<<<dim3(32, 6), 32, 0, stream>>>(Wq, Wk, Wv, Wo, W1, W2, frag);

    // 2) Q/K/V projections: (32768 x 128) @ (128 x 128)
    gemm128_kernel<<<MTOT / 16, 256, 0, stream>>>(X, fWq, bq, Qout, 0);
    gemm128_kernel<<<MTOT / 16, 256, 0, stream>>>(X, fWk, bk, Kout, 0);
    gemm128_kernel<<<MTOT / 16, 256, 0, stream>>>(X, fWv, bv, Vout, 0);

    // 3) attention: one wave per vertex, 8 waves per 256-thread block
    attn_kernel<<<MTOT / 8, 256, 0, stream>>>(Qout, Kout, Vout, nidx, vlen, Ctx);

    // 4) output projection + FFN
    gemm128_kernel<<<MTOT / 16, 256, 0, stream>>>(Ctx,  fWo, bo, buf1, 0);
    gemm128_kernel<<<MTOT / 16, 256, 0, stream>>>(buf1, fW1, b1, buf2, 1);
    gemm128_kernel<<<MTOT / 16, 256, 0, stream>>>(buf2, fW2, b2, (float*)d_out, 0);
}
